// PointCloudEncoder_19799799234730
// MI455X (gfx1250) — matrix operand order repaired and checked by reading, not compiled, not run on hardware
//
#include <hip/hip_runtime.h>
#include <hip/hip_bf16.h>

typedef _Float16 v16h __attribute__((ext_vector_type(16)));
typedef _Float16 v8h  __attribute__((ext_vector_type(8)));
typedef float    v8f  __attribute__((ext_vector_type(8)));

#define BATCH 8
#define NPTS  8192
#define MROWS (BATCH * NPTS)
#define OUTC  1923
#define EPSF  1e-8f

union FragH { v16h v; v8h h[2]; };

// ---------------------------------------------------------------------------
// WMMA GEMM: D[M,Nc] = act( A[M,K](f16) * W[Nc,K](f16)^T + bias )  f32 accum
// A is row-major [M][K]; W is row-major [Nc][K] (reference weight layout).
// block = 256 threads (8 waves), tile 128x64, BK=32, wave tile 32x32 (2x2 wmma)
// Double-buffered LDS: one barrier per K-step, staging overlaps WMMA.
// ---------------------------------------------------------------------------
__global__ __launch_bounds__(256)
void gemm_wmma_kernel(const _Float16* __restrict__ A, int lda,
                      const _Float16* __restrict__ W, int ldw,
                      const float* __restrict__ bias,
                      _Float16* __restrict__ Dh, int ldh,
                      float* __restrict__ Df, int ldf,
                      int K, int relu)
{
    __shared__ _Float16 As [2][128][32];   // A tile, row-major
    __shared__ _Float16 Bst[2][64][32];    // B tile as [n][k] (W native layout)

    const int t    = threadIdx.x;
    const int lane = t & 31;
    const int wid  = t >> 5;
    const int wr   = wid >> 1;             // 0..3  (M direction)
    const int wc   = wid & 1;              // 0..1  (N direction)
    const int blockRow = blockIdx.x * 128;
    const int blockCol = blockIdx.y * 64;

    v8f acc[2][2];
    v8f zro = {};
    acc[0][0] = zro; acc[0][1] = zro; acc[1][0] = zro; acc[1][1] = zro;

    // staging coordinates (per thread)
    const int arow = t >> 1;               // 0..127
    const int acol = (t & 1) * 16;         // 0 or 16 halves
    const int brow = t >> 2;               // 0..63
    const int bcol = (t & 3) * 8;          // 0,8,16,24 halves

    const _Float16* srcA = A + (size_t)(blockRow + arow) * lda + acol;
    const _Float16* srcB = W + (size_t)(blockCol + brow) * ldw + bcol;

    // fragment coordinates (per lane)
    const int m0  = lane & 15;
    const int kb  = (lane >> 4) * 8;       // A half-K base within [0,32)
    const int kb2 = (lane >> 4) * 8;       // B K base within [0,32)

    // ---- prologue: stage K-step 0 into buffer 0 ----
    {
        ((uint4*)&As[0][arow][acol])[0] = ((const uint4*)srcA)[0];
        ((uint4*)&As[0][arow][acol])[1] = ((const uint4*)srcA)[1];
        *(v8h*)&Bst[0][brow][bcol] = *(const v8h*)srcB;
    }

    int buf = 0;
    for (int k0 = 0; k0 < K; k0 += 32, buf ^= 1) {
        __syncthreads();   // staged tile visible; prior reads of other buf done

        // ---- stage next K-step into the other buffer (overlaps WMMA) ----
        if (k0 + 32 < K) {
            const _Float16* nA = srcA + k0 + 32;
            const _Float16* nB = srcB + k0 + 32;
            ((uint4*)&As[buf ^ 1][arow][acol])[0] = ((const uint4*)nA)[0];
            ((uint4*)&As[buf ^ 1][arow][acol])[1] = ((const uint4*)nA)[1];
            *(v8h*)&Bst[buf ^ 1][brow][bcol] = *(const v8h*)nB;
            if (k0 + 64 < K)
                __builtin_prefetch(nA + 32, 0, 1);   // global_prefetch_b8
        }

        // ---- fragments (ISA 16-bit A layout / same k order for B) ----
        FragH fa[2], fb[2];
        fa[0].h[0] = *(const v8h*)&As[buf][wr * 32 +      m0][kb];
        fa[0].h[1] = *(const v8h*)&As[buf][wr * 32 +      m0][kb + 16];
        fa[1].h[0] = *(const v8h*)&As[buf][wr * 32 + 16 + m0][kb];
        fa[1].h[1] = *(const v8h*)&As[buf][wr * 32 + 16 + m0][kb + 16];
        fb[0].h[0] = *(const v8h*)&Bst[buf][wc * 32 +      m0][kb2];
        fb[0].h[1] = *(const v8h*)&Bst[buf][wc * 32 +      m0][kb2 + 16];
        fb[1].h[0] = *(const v8h*)&Bst[buf][wc * 32 + 16 + m0][kb2];
        fb[1].h[1] = *(const v8h*)&Bst[buf][wc * 32 + 16 + m0][kb2 + 16];

        #pragma unroll
        for (int i = 0; i < 2; ++i)
            #pragma unroll
            for (int j = 0; j < 2; ++j)
                acc[i][j] = __builtin_amdgcn_wmma_f32_16x16x32_f16(
                    false, fa[i].v, false, fb[j].v,
                    (short)0, acc[i][j], false, false);
    }

    // ---- epilogue: f32 C/D layout: VGPR r -> M = r + 8*(lane>=16), N = lane&15
    const int rofs = (lane >> 4) * 8;
    const int ncol = lane & 15;
    #pragma unroll
    for (int ti = 0; ti < 2; ++ti) {
        #pragma unroll
        for (int tj = 0; tj < 2; ++tj) {
            const int baseR = blockRow + wr * 32 + ti * 16 + rofs;
            const int baseC = blockCol + wc * 32 + tj * 16 + ncol;
            const float bv = bias[baseC];
            #pragma unroll
            for (int r = 0; r < 8; ++r) {
                float v = acc[ti][tj][r] + bv;
                if (relu) v = v > 0.f ? v : 0.f;
                if (Dh) Dh[(size_t)(baseR + r) * ldh + baseC] = (_Float16)v;
                if (Df) Df[(size_t)(baseR + r) * ldf + baseC] = v;
            }
        }
    }
}

// ---------------------------------------------------------------------------
// weight convert: f32 (co,ci) row-major -> f16 [co][kpad], zero padded K
// (keeps the reference layout; GEMM consumes it directly as B[n][k])
// ---------------------------------------------------------------------------
__global__ void wconv_kernel(const float* __restrict__ w, _Float16* __restrict__ wt,
                             int ci, int co, int kpad)
{
    int i = blockIdx.x * 256 + threadIdx.x;
    if (i >= kpad * co) return;
    int n = i / kpad, k = i - n * kpad;
    wt[i] = (k < ci) ? (_Float16)w[(size_t)n * ci + k] : (_Float16)0.f;
}

// points -> padded f16 X0 [M][32]
__global__ void x0_kernel(const float* __restrict__ pts, _Float16* __restrict__ x0)
{
    int i = blockIdx.x * 256 + threadIdx.x;
    if (i >= MROWS * 32) return;
    int m = i >> 5, k = i & 31;
    x0[i] = (k < 3) ? (_Float16)pts[(size_t)m * 3 + k] : (_Float16)0.f;
}

// per-(b,c) image mean and sqrt(var_ddof1 + eps)
__global__ __launch_bounds__(256)
void imgstats_kernel(const float* __restrict__ img, float* __restrict__ im,
                     float* __restrict__ sc, int HW)
{
    __shared__ float ss[256], qq[256];
    const float* p = img + (size_t)blockIdx.x * HW;
    float s = 0.f, q = 0.f;
    for (int i = threadIdx.x; i < HW; i += 256) { float v = p[i]; s += v; q += v * v; }
    ss[threadIdx.x] = s; qq[threadIdx.x] = q;
    __syncthreads();
    for (int st = 128; st; st >>= 1) {
        if (threadIdx.x < st) {
            ss[threadIdx.x] += ss[threadIdx.x + st];
            qq[threadIdx.x] += qq[threadIdx.x + st];
        }
        __syncthreads();
    }
    if (threadIdx.x == 0) {
        float mean = ss[0] / (float)HW;
        float var  = (qq[0] - (float)HW * mean * mean) / (float)(HW - 1);
        im[blockIdx.x] = mean;
        sc[blockIdx.x] = sqrtf(var + EPSF);
    }
}

// per-point normalize + combine with image stats -> f16 T  (one wave per row)
__global__ __launch_bounds__(256)
void rownorm_kernel(const _Float16* __restrict__ X, _Float16* __restrict__ T,
                    const float* __restrict__ im, const float* __restrict__ sc, int C)
{
    const int lane = threadIdx.x & 31;
    const size_t row = (size_t)blockIdx.x * 8 + (threadIdx.x >> 5);
    const int b = (int)(row >> 13);
    const _Float16* xr = X + row * C;
    float s = 0.f, q = 0.f;
    for (int c = lane; c < C; c += 32) { float v = (float)xr[c]; s += v; q += v * v; }
    for (int off = 16; off; off >>= 1) { s += __shfl_xor(s, off); q += __shfl_xor(q, off); }
    const float mu  = s / (float)C;
    const float var = (q - (float)C * mu * mu) / (float)(C - 1);
    const float inv = 1.f / sqrtf(var + EPSF);
    _Float16* tr = T + row * C;
    const float* imb = im + (size_t)b * C;
    const float* scb = sc + (size_t)b * C;
    for (int c = lane; c < C; c += 32) {
        float v = ((float)xr[c] - mu) * inv;
        tr[c] = (_Float16)((v + imb[c]) * scb[c]);
    }
}

// ---------------------------------------------------------------------------
// projection path
// ---------------------------------------------------------------------------
__device__ __forceinline__ unsigned f2ord(float f) {
    unsigned u = __float_as_uint(f);
    return (u & 0x80000000u) ? ~u : (u | 0x80000000u);
}
__device__ __forceinline__ float ord2f(unsigned u) {
    return __uint_as_float((u & 0x80000000u) ? (u & 0x7fffffffu) : ~u);
}

__global__ void zinit_kernel(unsigned* zmin, unsigned* zmax)
{
    int i = threadIdx.x;
    if (i < BATCH) { zmin[i] = 0xFFFFFFFFu; zmax[i] = 0u; }
}

__global__ __launch_bounds__(256)
void zminmax_kernel(const float* __restrict__ pts, const float* __restrict__ proj,
                    unsigned* __restrict__ zmin, unsigned* __restrict__ zmax)
{
    int i = blockIdx.x * 256 + threadIdx.x;
    int b = i >> 13;                       // warp-uniform (8192 % 32 == 0)
    const float* P = proj + (size_t)b * 16;
    float px = pts[(size_t)i * 3], py = pts[(size_t)i * 3 + 1], pz = pts[(size_t)i * 3 + 2];
    float c2 = P[8]  * px + P[9]  * py + P[10] * pz + P[11];
    float c3 = P[12] * px + P[13] * py + P[14] * pz + P[15];
    unsigned u = f2ord(c2 / c3);
    unsigned umin = u, umax = u;
    for (int off = 16; off; off >>= 1) {
        umin = min(umin, (unsigned)__shfl_xor((int)umin, off));
        umax = max(umax, (unsigned)__shfl_xor((int)umax, off));
    }
    if ((threadIdx.x & 31) == 0) {
        atomicMin(&zmin[b], umin);
        atomicMax(&zmax[b], umax);
    }
}

// one wave per point, lanes stride over channels
__global__ __launch_bounds__(256)
void project_kernel(const float* __restrict__ pts, const float* __restrict__ img,
                    const float* __restrict__ proj,
                    const unsigned* __restrict__ zminU, const unsigned* __restrict__ zmaxU,
                    float* __restrict__ out, int C, int H, int W, int chOff)
{
    const int lane = threadIdx.x & 31;
    const size_t pt = (size_t)blockIdx.x * 8 + (threadIdx.x >> 5);
    const int b = (int)(pt >> 13);
    const float px = pts[pt * 3], py = pts[pt * 3 + 1], pz = pts[pt * 3 + 2];
    const float* P = proj + (size_t)b * 16;
    float c0 = P[0]  * px + P[1]  * py + P[2]  * pz + P[3];
    float c1 = P[4]  * px + P[5]  * py + P[6]  * pz + P[7];
    float c2 = P[8]  * px + P[9]  * py + P[10] * pz + P[11];
    float c3 = P[12] * px + P[13] * py + P[14] * pz + P[15];
    float x = (c0 / c3 + 1.f) * 0.5f;
    float y = 1.f - (c1 / c3 + 1.f) * 0.5f;
    float xs = fminf(fmaxf(x * (float)W, 0.f), (float)(W - 1));
    float ys = fminf(fmaxf(y * (float)H, 0.f), (float)(H - 1));
    float x1 = floorf(xs), x2 = ceilf(xs);
    float y1 = floorf(ys), y2 = ceilf(ys);
    int xi1 = (int)x1, xi2 = (int)x2, yi1 = (int)y1, yi2 = (int)y2;
    float wx2 = x2 - xs, wx1 = xs - x1, wy2 = y2 - ys, wy1 = ys - y1;
    float z = c2 / c3;
    float zmn = ord2f(zminU[b]), zmx = ord2f(zmaxU[b]);
    float zw = __expf((zmn - z) / (zmx - zmn));
    const float* base = img + (size_t)b * C * H * W;
    float* orow = out + pt * OUTC + chOff;
    for (int c = lane; c < C; c += 32) {
        const float* ch = base + (size_t)c * H * W;
        float Q11 = ch[yi1 * W + xi1];
        float Q12 = ch[yi2 * W + xi1];
        float Q21 = ch[yi1 * W + xi2];
        float Q22 = ch[yi2 * W + xi2];
        orow[c] = zw * (wx2 * (wy2 * Q11 + wy1 * Q12) + wx1 * (wy2 * Q21 + wy1 * Q22));
    }
}

__global__ void copy_points_kernel(const float* __restrict__ pts, float* __restrict__ out)
{
    int i = blockIdx.x * 256 + threadIdx.x;
    if (i >= MROWS * 3) return;
    int m = i / 3, k = i - m * 3;
    out[(size_t)m * OUTC + 1920 + k] = pts[i];
}

// ---------------------------------------------------------------------------
extern "C" void kernel_launch(void* const* d_in, const int* in_sizes, int n_in,
                              void* d_out, int out_size, void* d_ws, size_t ws_size,
                              hipStream_t stream)
{
    (void)in_sizes; (void)n_in; (void)out_size; (void)ws_size;

    const float* points = (const float*)d_in[0];
    const float* imgs[4] = { (const float*)d_in[2], (const float*)d_in[3],
                             (const float*)d_in[4], (const float*)d_in[5] };
    const float* proj = (const float*)d_in[6];
    float* out = (float*)d_out;

    const int Cs [4] = { 64, 128, 256, 512 };   // stage output / image channels
    const int Hs [4] = { 56, 28, 14, 7 };       // image spatial
    const int CIs[4] = { 3, 64, 128, 256 };     // stage input dims
    const int KPs[4] = { 32, 64, 128, 256 };    // padded K for first linear
    const int tOff[4] = { 0, 64, 192, 448 };    // transform channel offsets
    const int pOff[4] = { 960, 1024, 1152, 1408 };

    // bump allocator on d_ws
    char* wsb = (char*)d_ws;
    size_t off = 0;
    auto alloc = [&](size_t bytes) -> void* {
        off = (off + 255) & ~(size_t)255;
        void* p = wsb + off;
        off += bytes;
        return p;
    };

    _Float16* P0 = (_Float16*)alloc((size_t)MROWS * 256 * sizeof(_Float16));
    _Float16* P1 = (_Float16*)alloc((size_t)MROWS * 512 * sizeof(_Float16));
    _Float16* P2 = (_Float16*)alloc((size_t)MROWS * 512 * sizeof(_Float16));
    _Float16* X0 = (_Float16*)alloc((size_t)MROWS * 32 * sizeof(_Float16));
    _Float16 *W1t[4], *W2t[4], *FCt[4];
    float *imL[4], *scL[4];
    for (int i = 0; i < 4; ++i) {
        W1t[i] = (_Float16*)alloc((size_t)Cs[i] * KPs[i] * sizeof(_Float16));
        W2t[i] = (_Float16*)alloc((size_t)Cs[i] * Cs[i] * sizeof(_Float16));
        FCt[i] = (_Float16*)alloc((size_t)Cs[i] * Cs[i] * sizeof(_Float16));
        imL[i] = (float*)alloc((size_t)BATCH * Cs[i] * sizeof(float));
        scL[i] = (float*)alloc((size_t)BATCH * Cs[i] * sizeof(float));
    }
    unsigned* zmin = (unsigned*)alloc(BATCH * sizeof(unsigned));
    unsigned* zmax = (unsigned*)alloc(BATCH * sizeof(unsigned));

    // ---- preprocessing: weights -> f16 [co][Kpad], image stats, padded X0 ----
    for (int i = 0; i < 4; ++i) {
        const float* w1  = (const float*)d_in[7 + i * 6 + 0];
        const float* w2  = (const float*)d_in[7 + i * 6 + 2];
        const float* fcw = (const float*)d_in[7 + i * 6 + 4];
        int n1 = Cs[i] * KPs[i];
        int n2 = Cs[i] * Cs[i];
        wconv_kernel<<<(n1 + 255) / 256, 256, 0, stream>>>(w1, W1t[i], CIs[i], Cs[i], KPs[i]);
        wconv_kernel<<<(n2 + 255) / 256, 256, 0, stream>>>(w2, W2t[i], Cs[i], Cs[i], Cs[i]);
        wconv_kernel<<<(n2 + 255) / 256, 256, 0, stream>>>(fcw, FCt[i], Cs[i], Cs[i], Cs[i]);
        imgstats_kernel<<<BATCH * Cs[i], 256, 0, stream>>>(imgs[i], imL[i], scL[i], Hs[i] * Hs[i]);
    }
    x0_kernel<<<(MROWS * 32 + 255) / 256, 256, 0, stream>>>(points, X0);

    // ---- MLP + transform pipeline (all GEMMs on WMMA) ----
    _Float16* inBuf = X0;
    int inLd = 32;
    _Float16* outBufs[4] = { P0, P2, P0, P2 };   // ping-pong, P1 is always mid/T
    for (int i = 0; i < 4; ++i) {
        const int co = Cs[i];
        const float* b1  = (const float*)d_in[7 + i * 6 + 1];
        const float* b2  = (const float*)d_in[7 + i * 6 + 3];
        const float* fcb = (const float*)d_in[7 + i * 6 + 5];
        dim3 grid(MROWS / 128, co / 64);
        // linear1 + relu  (K padded to KPs[i], zeros in both operands)
        gemm_wmma_kernel<<<grid, 256, 0, stream>>>(inBuf, inLd, W1t[i], KPs[i], b1,
                                                   P1, co, nullptr, 0, KPs[i], 1);
        // linear2 + relu -> pc_feat
        _Float16* ob = outBufs[i];
        gemm_wmma_kernel<<<grid, 256, 0, stream>>>(P1, co, W2t[i], co, b2,
                                                   ob, co, nullptr, 0, co, 1);
        // normalize + image stats -> T (in P1)
        rownorm_kernel<<<MROWS / 8, 256, 0, stream>>>(ob, P1, imL[i], scL[i], co);
        // fc linear -> strided f32 slice of output
        gemm_wmma_kernel<<<grid, 256, 0, stream>>>(P1, co, FCt[i], co, fcb,
                                                   nullptr, 0, out + tOff[i], OUTC, co, 0);
        inBuf = ob; inLd = co;
    }

    // ---- projection path ----
    zinit_kernel<<<1, BATCH, 0, stream>>>(zmin, zmax);
    zminmax_kernel<<<MROWS / 256, 256, 0, stream>>>(points, proj, zmin, zmax);
    for (int i = 0; i < 4; ++i)
        project_kernel<<<MROWS / 8, 256, 0, stream>>>(points, imgs[i], proj, zmin, zmax,
                                                      out, Cs[i], Hs[i], Hs[i], pOff[i]);
    copy_points_kernel<<<(MROWS * 3 + 255) / 256, 256, 0, stream>>>(points, out);
}
